// FlashFWSVDBlock_1468878815661
// MI455X (gfx1250) — compile-verified, matrix operand order checked by reading, not run on hardware
//
#include <hip/hip_runtime.h>
#include <hip/hip_bf16.h>
#include <math.h>

// ---------------------------------------------------------------------------
// Dims (fixed by the reference)
// ---------------------------------------------------------------------------
#define BB   2
#define MM_  2048
#define DM_  1024
#define HH   16
#define DH_  64
#define RR   32
#define RFF_ 256
#define RWO_ 256
#define DFF_ 4096

typedef __attribute__((ext_vector_type(16))) _Float16 v16h;
typedef __attribute__((ext_vector_type(8)))  _Float16 h8;
typedef __attribute__((ext_vector_type(8)))  float    v8f;

#define DEVINL __device__ __forceinline__
#define AS3 __attribute__((address_space(3)))

// ---- CDNA5 async global->LDS copy (VGLOBAL GLOBAL_LOAD_ASYNC_TO_LDS_B128,
// tracked by ASYNCcnt).  Inline asm: the clang builtin on this toolchain is a
// buffer-descriptor variant (v4i V# first arg), so we emit the GV-mode
// instruction directly.  Offset imm applies to both LDS and global addresses
// (ISA 15.18.3), so one base pair covers 64B per lane.
// ---------------------------------------------------------------------------
#if defined(__gfx1250__)
#define HAS_ASYNC_ASM 1
#endif

DEVINL void async_copy64B(const _Float16* g, _Float16* l) {
#if defined(HAS_ASYNC_ASM)
  unsigned lo = (unsigned)(size_t)(AS3 _Float16*)l;
  unsigned long long ga = (unsigned long long)(size_t)g;
  asm volatile(
      "global_load_async_to_lds_b128 %0, %1, off\n\t"
      "global_load_async_to_lds_b128 %0, %1, off offset:16\n\t"
      "global_load_async_to_lds_b128 %0, %1, off offset:32\n\t"
      "global_load_async_to_lds_b128 %0, %1, off offset:48"
      :: "v"(lo), "v"(ga) : "memory");
#else
  const h8* s = (const h8*)g;
  h8* d = (h8*)l;
  d[0] = s[0]; d[1] = s[1]; d[2] = s[2]; d[3] = s[3];
#endif
}

DEVINL void async_copy32B(const _Float16* g, _Float16* l) {
#if defined(HAS_ASYNC_ASM)
  unsigned lo = (unsigned)(size_t)(AS3 _Float16*)l;
  unsigned long long ga = (unsigned long long)(size_t)g;
  asm volatile(
      "global_load_async_to_lds_b128 %0, %1, off\n\t"
      "global_load_async_to_lds_b128 %0, %1, off offset:16"
      :: "v"(lo), "v"(ga) : "memory");
#else
  const h8* s = (const h8*)g;
  h8* d = (h8*)l;
  d[0] = s[0]; d[1] = s[1];
#endif
}

DEVINL void wait_async0() {
#if defined(HAS_ASYNC_ASM)
#if defined(__has_builtin) && __has_builtin(__builtin_amdgcn_s_wait_asynccnt)
  __builtin_amdgcn_s_wait_asynccnt(0);
#else
  asm volatile("s_wait_asynccnt 0x0" ::: "memory");
#endif
#endif
}

DEVINL void wait_async8() {
#if defined(HAS_ASYNC_ASM)
#if defined(__has_builtin) && __has_builtin(__builtin_amdgcn_s_wait_asynccnt)
  __builtin_amdgcn_s_wait_asynccnt(8);
#else
  asm volatile("s_wait_asynccnt 0x8" ::: "memory");
#endif
#endif
}

// D = A(16x32 f16) * B(32x16 f16) + C(16x16 f32)
DEVINL v8f wmma32(v16h a, v16h b, v8f c) {
  return __builtin_amdgcn_wmma_f32_16x16x32_f16(false, a, false, b, (short)0, c,
                                                false, false);
}

// Load a 16x32 f16 A-fragment (also used for B^T fragments) from a row-major
// matrix with leading dimension `ld` (halves). CDNA5 16-bit A layout:
//   lanes 0-15 : row = lane,    elements 0..7 = K 0..7,  8..15 = K 16..23
//   lanes 16-31: row = lane-16, elements 0..7 = K 8..15, 8..15 = K 24..31
DEVINL v16h load_frag(const _Float16* base, int ld, int row0, int k0) {
  const int lane = threadIdx.x & 31;
  const int r    = lane & 15;
  const int koff = (lane & 16) ? 8 : 0;
  const _Float16* p = base + (size_t)(row0 + r) * ld + (k0 + koff);
  union { v16h v; h8 h[2]; } u;
  u.h[0] = *(const h8*)(p);
  u.h[1] = *(const h8*)(p + 16);
  return u.v;
}

// ---------------------------------------------------------------------------
// Small prep kernels
// ---------------------------------------------------------------------------
__global__ void conv_f16_kernel(const float* __restrict__ in,
                                _Float16* __restrict__ out, int n) {
  int id = blockIdx.x * 256 + threadIdx.x;
  if (id < n) out[id] = (_Float16)in[id];
}

// in [K,N] f32  ->  out [N,K] f16   (weights stored transposed for WMMA B)
__global__ void tconv_kernel(const float* __restrict__ in,
                             _Float16* __restrict__ out, int K, int N) {
  int id = blockIdx.x * 256 + threadIdx.x;
  if (id >= K * N) return;
  int k = id / N, n = id - k * N;
  out[(size_t)n * K + k] = (_Float16)in[id];
}

// W^T[n= h*64+dh][dm] = scale * sum_r P[h][dm][r] * V[h][r][dh];  bias scaled too
__global__ void build_w_kernel(const float* __restrict__ P,
                               const float* __restrict__ V,
                               const float* __restrict__ bin, float scale,
                               _Float16* __restrict__ Wt,
                               float* __restrict__ bout) {
  int id = blockIdx.x * 256 + threadIdx.x;   // over 1024*1024
  int n = id >> 10, dm = id & 1023;
  int h = n >> 6, dh = n & 63;
  const float* Pp = P + ((size_t)h * DM_ + dm) * RR;
  const float* Vp = V + ((size_t)h * RR) * DH_ + dh;
  float s = 0.f;
#pragma unroll
  for (int r = 0; r < RR; ++r) s += Pp[r] * Vp[(size_t)r * DH_];
  Wt[(size_t)n * DM_ + dm] = (_Float16)(s * scale);
  if (dm == 0) bout[n] = bin[n] * scale;
}

// ---------------------------------------------------------------------------
// Generic f16 WMMA GEMM:  C[M,N] = A[M,K] * Bt[N,K]^T  (+bias, +gelu)
// 256 threads, tile 128x128, K-step 64, double-buffered async LDS staging;
// wave grid 2(M) x 4(N); wave tile 64x32.
// mode 0: f32 [M,N]   1: f16 [M,N]   2: f16 [B,H,M,DH]   3: f16 [B,H,DH,M]
// ---------------------------------------------------------------------------
__global__ __launch_bounds__(256) void gemm_kernel(
    const _Float16* __restrict__ A, const _Float16* __restrict__ Bt,
    const float* __restrict__ bias, void* __restrict__ out,
    int M, int N, int K, int mode, int act) {
  __shared__ __align__(16) _Float16 ldsA[2][128 * 64];
  __shared__ __align__(16) _Float16 ldsB[2][128 * 64];
  const int t = threadIdx.x;
  const int lane = t & 31;
  const int w = t >> 5;
  const int wm = w >> 2, wn = w & 3;
  const int m0 = blockIdx.y * 128;
  const int n0 = blockIdx.x * 128;

  v8f c[4][2];
#pragma unroll
  for (int mi = 0; mi < 4; ++mi)
#pragma unroll
    for (int ni = 0; ni < 2; ++ni)
#pragma unroll
      for (int e = 0; e < 8; ++e) c[mi][ni][e] = 0.f;

  const int lrow = t >> 1;          // 0..127
  const int lcol = (t & 1) * 32;    // 0 or 32

  // issue one 128x64 tile (one 64B async block per thread per matrix = 4
  // wave-level async instructions per matrix)
  auto issueA = [&](int buf, int k0) {
    async_copy64B(A + (size_t)(m0 + lrow) * K + k0 + lcol,
                  &ldsA[buf][lrow * 64 + lcol]);
  };
  auto issueB = [&](int buf, int k0) {
    async_copy64B(Bt + (size_t)(n0 + lrow) * K + k0 + lcol,
                  &ldsB[buf][lrow * 64 + lcol]);
  };

  const int nk = K >> 6;
  issueA(0, 0);
  issueB(0, 0);
  for (int ki = 0; ki < nk; ++ki) {
    const int buf = ki & 1;
    if (ki + 1 < nk) {
      issueA(buf ^ 1, (ki + 1) << 6);
      issueB(buf ^ 1, (ki + 1) << 6);
      wait_async8();   // current tile done (async completes in order)
    } else {
      wait_async0();
    }
    __syncthreads();
#pragma unroll
    for (int kk = 0; kk < 64; kk += 32) {
      v16h a[4], bf[2];
#pragma unroll
      for (int mi = 0; mi < 4; ++mi)
        a[mi] = load_frag(ldsA[buf], 64, wm * 64 + mi * 16, kk);
#pragma unroll
      for (int ni = 0; ni < 2; ++ni)
        bf[ni] = load_frag(ldsB[buf], 64, wn * 32 + ni * 16, kk);
#pragma unroll
      for (int mi = 0; mi < 4; ++mi)
#pragma unroll
        for (int ni = 0; ni < 2; ++ni)
          c[mi][ni] = wmma32(a[mi], bf[ni], c[mi][ni]);
    }
    __syncthreads();   // all waves done reading buf before it is re-filled
  }

  // Epilogue. D layout: VGPR i -> row = i + 8*(lane>>4); col = lane&15.
  const int rhalf = (lane >> 4) << 3;
  const int cl = lane & 15;
#pragma unroll
  for (int mi = 0; mi < 4; ++mi)
#pragma unroll
    for (int ni = 0; ni < 2; ++ni) {
      const int rbase = m0 + wm * 64 + mi * 16 + rhalf;
      const int gcol = n0 + wn * 32 + ni * 16 + cl;
#pragma unroll
      for (int i = 0; i < 8; ++i) {
        const int grow = rbase + i;
        float acc = c[mi][ni][i];
        if (bias) acc += bias[gcol];
        if (act == 1)  // exact gelu
          acc = 0.5f * acc * (1.f + erff(acc * 0.70710678118654752f));
        if (mode == 0) {
          ((float*)out)[(size_t)grow * N + gcol] = acc;
        } else if (mode == 1) {
          ((_Float16*)out)[(size_t)grow * N + gcol] = (_Float16)acc;
        } else {
          const int b = grow >> 11, m = grow & 2047;
          const int h = gcol >> 6, dh = gcol & 63;
          if (mode == 2)
            ((_Float16*)out)[((((size_t)b * HH + h) * MM_) + m) * DH_ + dh] =
                (_Float16)acc;
          else
            ((_Float16*)out)[((((size_t)b * HH + h) * DH_) + dh) * MM_ + m] =
                (_Float16)acc;
        }
      }
    }
}

// ---------------------------------------------------------------------------
// Flash attention: grid (M/128, B*H); 8 waves x 16 Q-rows; KV tiles of 64.
// q,k: [B,H,M,DH] f16 (q pre-scaled by 1/sqrt(DH)); v: [B,H,DH,M] f16.
// out: attn [B*M, DM] f16 with col = h*64+dh (matches reference reshape).
// ---------------------------------------------------------------------------
__global__ __launch_bounds__(256) void flash_kernel(
    const _Float16* __restrict__ qf, const _Float16* __restrict__ kf,
    const _Float16* __restrict__ vtf, _Float16* __restrict__ attn) {
  __shared__ __align__(16) _Float16 ldsK[64 * 64];        // [kv][dh]
  __shared__ __align__(16) _Float16 ldsV[64 * 64];        // [dh][kv]
  __shared__ __align__(16) _Float16 ldsP[8 * 16 * 64];    // per-wave P bounce
  const int t = threadIdx.x;
  const int lane = t & 31;
  const int w = t >> 5;
  const int bh = blockIdx.y;
  const int b = bh >> 4, h = bh & 15;
  const int m0 = blockIdx.x * 128;
  const int qrow0 = m0 + w * 16;

  const _Float16* qb = qf + (size_t)bh * MM_ * DH_;
  const _Float16* kb = kf + (size_t)bh * MM_ * DH_;
  const _Float16* vb = vtf + (size_t)bh * DH_ * MM_;

  const v16h aq0 = load_frag(qb, DH_, qrow0, 0);
  const v16h aq1 = load_frag(qb, DH_, qrow0, 32);

  v8f o[4];
  float mrow[8], lrow[8];
#pragma unroll
  for (int j = 0; j < 4; ++j)
#pragma unroll
    for (int e = 0; e < 8; ++e) o[j][e] = 0.f;
#pragma unroll
  for (int i = 0; i < 8; ++i) { mrow[i] = -3.0e38f; lrow[i] = 0.f; }

  const int rhalf = (lane >> 4) << 3;
  const int cl = lane & 15;
  _Float16* pw = ldsP + w * (16 * 64);

  const int vr = t >> 2, vc = (t & 3) * 16;   // V^T tile copy mapping

  for (int kv0 = 0; kv0 < MM_; kv0 += 64) {
    // stage K tile (contiguous rows) and V^T tile (rows stride M) via async DMA
    async_copy32B(kb + (size_t)kv0 * DH_ + t * 16, ldsK + t * 16);
    async_copy32B(vb + (size_t)vr * MM_ + kv0 + vc, ldsV + vr * 64 + vc);
    wait_async0();
    __syncthreads();

    // S = Q * K^T  (4 n-tiles of 16 kv columns)
    v8f s[4];
#pragma unroll
    for (int j = 0; j < 4; ++j) {
      v8f z;
#pragma unroll
      for (int e = 0; e < 8; ++e) z[e] = 0.f;
      v16h bk0 = load_frag(ldsK, 64, j * 16, 0);
      v16h bk1 = load_frag(ldsK, 64, j * 16, 32);
      z = wmma32(aq0, bk0, z);
      s[j] = wmma32(aq1, bk1, z);
    }

    // online softmax: row stats via 16-lane butterflies (rows stay in halves)
    float mnew[8], alpha[8];
#pragma unroll
    for (int i = 0; i < 8; ++i) {
      float tmax = fmaxf(fmaxf(s[0][i], s[1][i]), fmaxf(s[2][i], s[3][i]));
      tmax = fmaxf(tmax, __shfl_xor(tmax, 1));
      tmax = fmaxf(tmax, __shfl_xor(tmax, 2));
      tmax = fmaxf(tmax, __shfl_xor(tmax, 4));
      tmax = fmaxf(tmax, __shfl_xor(tmax, 8));
      mnew[i] = fmaxf(mrow[i], tmax);
      alpha[i] = __expf(mrow[i] - mnew[i]);
      mrow[i] = mnew[i];
    }
    float rs[8];
#pragma unroll
    for (int i = 0; i < 8; ++i) rs[i] = 0.f;
#pragma unroll
    for (int j = 0; j < 4; ++j)
#pragma unroll
      for (int i = 0; i < 8; ++i) {
        float p = __expf(s[j][i] - mnew[i]);
        s[j][i] = p;
        rs[i] += p;
      }
#pragma unroll
    for (int i = 0; i < 8; ++i) {
      float r = rs[i];
      r += __shfl_xor(r, 1);
      r += __shfl_xor(r, 2);
      r += __shfl_xor(r, 4);
      r += __shfl_xor(r, 8);
      lrow[i] = lrow[i] * alpha[i] + r;
    }
#pragma unroll
    for (int j = 0; j < 4; ++j)
#pragma unroll
      for (int i = 0; i < 8; ++i) o[j][i] *= alpha[i];

    // D-layout -> A-layout relayout of P through per-wave LDS (same-wave DS
    // ops are in-order; no workgroup barrier needed here)
#pragma unroll
    for (int j = 0; j < 4; ++j)
#pragma unroll
      for (int i = 0; i < 8; ++i)
        pw[(i + rhalf) * 64 + j * 16 + cl] = (_Float16)s[j][i];
    v16h ap0 = load_frag(pw, 64, 0, 0);
    v16h ap1 = load_frag(pw, 64, 0, 32);

    // O += P * V  (B-frags from V^T tile, contiguous in kv)
#pragma unroll
    for (int j = 0; j < 4; ++j) {
      v16h bv0 = load_frag(ldsV, 64, j * 16, 0);
      v16h bv1 = load_frag(ldsV, 64, j * 16, 32);
      o[j] = wmma32(ap0, bv0, o[j]);
      o[j] = wmma32(ap1, bv1, o[j]);
    }
    __syncthreads();
  }

  // normalize and store as [B*M, DM] f16, col = h*64 + dh
#pragma unroll
  for (int j = 0; j < 4; ++j)
#pragma unroll
    for (int i = 0; i < 8; ++i) {
      float val = o[j][i] / lrow[i];
      const int grow = qrow0 + rhalf + i;
      const int gcol = h * DH_ + j * 16 + cl;
      attn[((size_t)(b * MM_ + grow)) * DM_ + gcol] = (_Float16)val;
    }
}

// ---------------------------------------------------------------------------
// LayerNorm over rows of 1024:  out = LN(a + bsrc + bias) * g + beta
// ---------------------------------------------------------------------------
__global__ __launch_bounds__(256) void ln_kernel(
    const float* __restrict__ a, const float* __restrict__ bsrc,
    const float* __restrict__ bias, const float* __restrict__ g,
    const float* __restrict__ beta, float* __restrict__ out32,
    _Float16* __restrict__ out16) {
  __shared__ float red[256];
  const int row = blockIdx.x;
  const int t = threadIdx.x;
  float v[4];
  float s = 0.f;
#pragma unroll
  for (int i = 0; i < 4; ++i) {
    const int col = i * 256 + t;
    float x = a[(size_t)row * DM_ + col] + bsrc[(size_t)row * DM_ + col];
    if (bias) x += bias[col];
    v[i] = x;
    s += x;
  }
  red[t] = s;
  __syncthreads();
  for (int off = 128; off > 0; off >>= 1) {
    if (t < off) red[t] += red[t + off];
    __syncthreads();
  }
  const float mu = red[0] * (1.f / DM_);
  __syncthreads();
  float vs = 0.f;
#pragma unroll
  for (int i = 0; i < 4; ++i) {
    const float d = v[i] - mu;
    vs += d * d;
  }
  red[t] = vs;
  __syncthreads();
  for (int off = 128; off > 0; off >>= 1) {
    if (t < off) red[t] += red[t + off];
    __syncthreads();
  }
  const float rstd = rsqrtf(red[0] * (1.f / DM_) + 1e-12f);
#pragma unroll
  for (int i = 0; i < 4; ++i) {
    const int col = i * 256 + t;
    const float y = (v[i] - mu) * rstd * g[col] + beta[col];
    out32[(size_t)row * DM_ + col] = y;
    if (out16) out16[(size_t)row * DM_ + col] = (_Float16)y;
  }
}

// ---------------------------------------------------------------------------
// Host: pipeline
// ---------------------------------------------------------------------------
extern "C" void kernel_launch(void* const* d_in, const int* in_sizes, int n_in,
                              void* d_out, int out_size, void* d_ws,
                              size_t ws_size, hipStream_t stream) {
  (void)in_sizes; (void)n_in; (void)out_size; (void)ws_size;
  const float* x     = (const float*)d_in[0];
  const float* Pq    = (const float*)d_in[1];
  const float* Vq    = (const float*)d_in[2];
  const float* bq    = (const float*)d_in[3];
  const float* Pk    = (const float*)d_in[4];
  const float* Vk    = (const float*)d_in[5];
  const float* bk    = (const float*)d_in[6];
  const float* Pv    = (const float*)d_in[7];
  const float* Vv    = (const float*)d_in[8];
  const float* bv    = (const float*)d_in[9];
  const float* Uo    = (const float*)d_in[10];
  const float* Vo    = (const float*)d_in[11];
  const float* bo    = (const float*)d_in[12];
  const float* U1    = (const float*)d_in[13];
  const float* V1    = (const float*)d_in[14];
  const float* b1    = (const float*)d_in[15];
  const float* U2    = (const float*)d_in[16];
  const float* V2    = (const float*)d_in[17];
  const float* b2    = (const float*)d_in[18];
  const float* ln1g  = (const float*)d_in[19];
  const float* ln1b  = (const float*)d_in[20];
  const float* ln2g  = (const float*)d_in[21];
  const float* ln2b  = (const float*)d_in[22];
  float* out = (float*)d_out;

  const size_t BM = (size_t)BB * MM_;  // 4096

  size_t off = 0;
  auto alloc = [&](size_t bytes) -> void* {
    void* r = (char*)d_ws + off;
    off = (off + bytes + 255) & ~(size_t)255;
    return r;
  };
  _Float16* xh   = (_Float16*)alloc(BM * DM_ * 2);
  _Float16* wqt  = (_Float16*)alloc((size_t)DM_ * DM_ * 2);
  _Float16* wkt  = (_Float16*)alloc((size_t)DM_ * DM_ * 2);
  _Float16* wvt  = (_Float16*)alloc((size_t)DM_ * DM_ * 2);
  float*    bqs  = (float*)alloc(DM_ * 4);
  float*    bks  = (float*)alloc(DM_ * 4);
  float*    bvs  = (float*)alloc(DM_ * 4);
  _Float16* uot  = (_Float16*)alloc((size_t)RWO_ * DM_ * 2);
  _Float16* vot  = (_Float16*)alloc((size_t)DM_ * RWO_ * 2);
  _Float16* u1t  = (_Float16*)alloc((size_t)RFF_ * DM_ * 2);
  _Float16* v1t  = (_Float16*)alloc((size_t)DFF_ * RFF_ * 2);
  _Float16* u2t  = (_Float16*)alloc((size_t)RFF_ * DFF_ * 2);
  _Float16* v2t  = (_Float16*)alloc((size_t)DM_ * RFF_ * 2);
  _Float16* qf   = (_Float16*)alloc(BM * DM_ * 2);
  _Float16* kfb  = (_Float16*)alloc(BM * DM_ * 2);
  _Float16* vtf  = (_Float16*)alloc(BM * DM_ * 2);
  _Float16* attn = (_Float16*)alloc(BM * DM_ * 2);
  _Float16* t0   = (_Float16*)alloc(BM * RWO_ * 2);
  float*    y0   = (float*)alloc(BM * DM_ * 4);
  float*    x1f  = (float*)alloc(BM * DM_ * 4);
  _Float16* x1h  = (_Float16*)alloc(BM * DM_ * 2);
  _Float16* t2   = (_Float16*)alloc(BM * RFF_ * 2);
  _Float16* mid  = (_Float16*)alloc(BM * (size_t)DFF_ * 2);
  _Float16* t4   = (_Float16*)alloc(BM * RFF_ * 2);
  float*    y1   = (float*)alloc(BM * DM_ * 4);

  const float qscale = 0.125f;  // 1/sqrt(64)

  // x -> f16
  conv_f16_kernel<<<(BM * DM_) / 256, 256, 0, stream>>>(x, xh, (int)(BM * DM_));

  // fold low-rank QKV into full per-head weights (transposed), scale folded
  build_w_kernel<<<(DM_ * DM_) / 256, 256, 0, stream>>>(Pq, Vq, bq, qscale, wqt, bqs);
  build_w_kernel<<<(DM_ * DM_) / 256, 256, 0, stream>>>(Pk, Vk, bk, 1.0f, wkt, bks);
  build_w_kernel<<<(DM_ * DM_) / 256, 256, 0, stream>>>(Pv, Vv, bv, 1.0f, wvt, bvs);

  // transpose+convert the remaining weights
  tconv_kernel<<<(DM_ * RWO_) / 256, 256, 0, stream>>>(Uo, uot, DM_, RWO_);
  tconv_kernel<<<(RWO_ * DM_) / 256, 256, 0, stream>>>(Vo, vot, RWO_, DM_);
  tconv_kernel<<<(DM_ * RFF_) / 256, 256, 0, stream>>>(U1, u1t, DM_, RFF_);
  tconv_kernel<<<(RFF_ * DFF_) / 256, 256, 0, stream>>>(V1, v1t, RFF_, DFF_);
  tconv_kernel<<<(DFF_ * RFF_) / 256, 256, 0, stream>>>(U2, u2t, DFF_, RFF_);
  tconv_kernel<<<(RFF_ * DM_) / 256, 256, 0, stream>>>(V2, v2t, RFF_, DM_);

  auto gemm = [&](const _Float16* A, const _Float16* Bt, const float* bias,
                  void* o, int M, int N, int K, int mode, int act) {
    dim3 grid(N / 128, M / 128);
    gemm_kernel<<<grid, 256, 0, stream>>>(A, Bt, bias, o, M, N, K, mode, act);
  };

  // Q/K [B,H,M,DH], V^T [B,H,DH,M]
  gemm(xh, wqt, bqs, qf,  (int)BM, DM_, DM_, 2, 0);
  gemm(xh, wkt, bks, kfb, (int)BM, DM_, DM_, 2, 0);
  gemm(xh, wvt, bvs, vtf, (int)BM, DM_, DM_, 3, 0);

  // fused flash attention -> attn [B*M, DM] f16
  {
    dim3 grid(MM_ / 128, BB * HH);
    flash_kernel<<<grid, 256, 0, stream>>>(qf, kfb, vtf, attn);
  }

  // attention output projection (low-rank) + residual + LN1
  gemm(attn, uot, nullptr, t0, (int)BM, RWO_, DM_, 1, 0);
  gemm(t0, vot, nullptr, y0, (int)BM, DM_, RWO_, 0, 0);
  ln_kernel<<<BM, 256, 0, stream>>>(x, y0, bo, ln1g, ln1b, x1f, x1h);

  // FFN (low-rank) with fused bias+gelu, then LN2 -> out
  gemm(x1h, u1t, nullptr, t2, (int)BM, RFF_, DM_, 1, 0);
  gemm(t2, v1t, b1, mid, (int)BM, DFF_, RFF_, 1, 1);
  gemm(mid, u2t, nullptr, t4, (int)BM, RFF_, DFF_, 1, 0);
  gemm(t4, v2t, b2, y1, (int)BM, DM_, RFF_, 0, 0);
  ln_kernel<<<BM, 256, 0, stream>>>(x1f, y1, nullptr, ln2g, ln2b, out, nullptr);
}
